// DSA_32255204393145
// MI455X (gfx1250) — matrix operand order repaired and checked by reading, not compiled, not run on hardware
//
#include <hip/hip_runtime.h>
#include <hip/hip_bf16.h>

typedef __attribute__((ext_vector_type(16))) _Float16 v16h;
typedef __attribute__((ext_vector_type(8)))  _Float16 v8h;
typedef __attribute__((ext_vector_type(8)))  float    v8f;
typedef __attribute__((ext_vector_type(4)))  float    v4f;

#define B_   2
#define S_   4096
#define DM   1024
#define HQ   8
#define DK   64
#define TOPK 256

// ---------------- Kernel 1: indexer projections + layernorm -> f16 ----------
// block = 128 threads, handles 8 rows of x. thread t computes output column t
// (0..63 -> q proj, 64..127 -> k proj) for all 8 rows, streaming one W row.
__global__ void __launch_bounds__(128)
proj_ln_kernel(const float* __restrict__ x,
               const float* __restrict__ Wq, const float* __restrict__ bq,
               const float* __restrict__ Wk, const float* __restrict__ bk,
               const float* __restrict__ g,  const float* __restrict__ bln,
               _Float16* __restrict__ qi, _Float16* __restrict__ ki) {
  __shared__ float xs[8][DM];
  const int tid  = threadIdx.x;
  const int row0 = blockIdx.x * 8;            // flat (b*S + s) row
  for (int i = tid; i < 8 * DM / 4; i += 128)
    ((v4f*)&xs[0][0])[i] = ((const v4f*)(x + (size_t)row0 * DM))[i];
  __syncthreads();

  const int  co   = tid;                      // output column 0..127
  const bool isK  = co >= 64;
  const int  wrow = co & 63;                  // h*32 + d
  const float* W  = (isK ? Wk : Wq) + (size_t)wrow * DM;
  const float bias = isK ? bk[wrow] : bq[wrow];

  float acc[8];
#pragma unroll
  for (int r = 0; r < 8; ++r) acc[r] = bias;
  for (int j = 0; j < DM; j += 4) {
    v4f w4 = *(const v4f*)(W + j);
#pragma unroll
    for (int r = 0; r < 8; ++r)
      acc[r] += w4.x * xs[r][j] + w4.y * xs[r][j + 1] +
                w4.z * xs[r][j + 2] + w4.w * xs[r][j + 3];
  }
  // layernorm over groups of 32 columns == one wave (block = 4 waves of 32)
  const int d  = co & 31;
  const float gg = g[d], bb = bln[d];
  _Float16* dst = isK ? ki : qi;
#pragma unroll
  for (int r = 0; r < 8; ++r) {
    float s = acc[r], s2 = acc[r] * acc[r];
#pragma unroll
    for (int m = 1; m < 32; m <<= 1) {
      s  += __shfl_xor(s,  m, 32);
      s2 += __shfl_xor(s2, m, 32);
    }
    const float mean = s * (1.0f / 32.0f);
    const float var  = s2 * (1.0f / 32.0f) - mean * mean;
    const float y = (acc[r] - mean) * rsqrtf(var + 1e-5f) * gg + bb;
    dst[(size_t)(row0 + r) * 64 + wrow] = (_Float16)y;
  }
}

// ---------------- Kernel 1b: one-time K/V fp32 -> f16 conversion ------------
__global__ void __launch_bounds__(256)
cvt_kv_kernel(const float* __restrict__ K, const float* __restrict__ V,
              _Float16* __restrict__ K16, _Float16* __restrict__ V16) {
  const size_t i = ((size_t)blockIdx.x * 256 + threadIdx.x) * 8;
  v4f a0 = *(const v4f*)(K + i);
  v4f a1 = *(const v4f*)(K + i + 4);
  v8h kh;
  kh[0] = (_Float16)a0.x; kh[1] = (_Float16)a0.y;
  kh[2] = (_Float16)a0.z; kh[3] = (_Float16)a0.w;
  kh[4] = (_Float16)a1.x; kh[5] = (_Float16)a1.y;
  kh[6] = (_Float16)a1.z; kh[7] = (_Float16)a1.w;
  *(v8h*)(K16 + i) = kh;
  v4f b0 = *(const v4f*)(V + i);
  v4f b1 = *(const v4f*)(V + i + 4);
  v8h vh;
  vh[0] = (_Float16)b0.x; vh[1] = (_Float16)b0.y;
  vh[2] = (_Float16)b0.z; vh[3] = (_Float16)b0.w;
  vh[4] = (_Float16)b1.x; vh[5] = (_Float16)b1.y;
  vh[6] = (_Float16)b1.z; vh[7] = (_Float16)b1.w;
  *(v8h*)(V16 + i) = vh;
}

// ---------------- Kernel 2: WMMA index scores (16 x 4096 in LDS) + top-k ----
#define SSTR 4100   // padded row stride (floats) for score stripe in LDS
static constexpr int SMEM_SCORE =
    16 * SSTR * 4 + 16 * 256 * 4 + 4 * 16 * 4;   // scores + hist + scalars

__global__ void __launch_bounds__(256)
score_topk_kernel(const _Float16* __restrict__ qi,
                  const _Float16* __restrict__ ki,
                  const float* __restrict__ idx_w,
                  int* __restrict__ top_idx) {
  extern __shared__ unsigned char smem[];
  float*    sc     = (float*)smem;                              // [16][SSTR]
  unsigned* hist   = (unsigned*)(smem + (size_t)16 * SSTR * 4); // [16][256]
  unsigned* prefix = hist + 16 * 256;                           // [16]
  int*      needed = (int*)(prefix + 16);                       // [16]
  int*      cntA   = needed + 16;                               // [16]
  int*      cntB   = cntA + 16;                                 // [16]

  const int tid  = threadIdx.x;
  const int lane = tid & 31;
  const int wave = tid >> 5;
  const int b    = blockIdx.x / (S_ / 16);
  const int s0   = (blockIdx.x % (S_ / 16)) * 16;
  const size_t rowbase = (size_t)b * S_;

  const float w0 = idx_w[0], w1 = idx_w[1];

  const int hiHalf = lane >> 4;     // lane half selects K sub-range
  const int mrow   = lane & 15;

  // A tiles: k_i rows s0..s0+15, heads 0 and 1 (16x32 f16)
  v16h a[2];
#pragma unroll
  for (int h = 0; h < 2; ++h) {
    const _Float16* p = ki + (rowbase + s0 + mrow) * 64 + h * 32 + hiHalf * 8;
    v8h lo = *(const v8h*)p;
    v8h hi = *(const v8h*)(p + 16);
#pragma unroll
    for (int i = 0; i < 8; ++i) { a[h][i] = lo[i]; a[h][8 + i] = hi[i]; }
  }

  // stream over 256 t-chunks; head_scores = relu(k_i . q_i), weighted sum
  for (int c = wave; c < 256; c += 8) {
    const int t0 = c * 16;
    v16h bm[2];
#pragma unroll
    for (int h = 0; h < 2; ++h) {
      const _Float16* p = qi + (rowbase + t0 + mrow) * 64 + h * 32 + hiHalf * 8;
      v8h lo = *(const v8h*)p;
      v8h hi = *(const v8h*)(p + 16);
#pragma unroll
      for (int i = 0; i < 8; ++i) { bm[h][i] = lo[i]; bm[h][8 + i] = hi[i]; }
    }
    v8f z = {};
    v8f d0 = __builtin_amdgcn_wmma_f32_16x16x32_f16(false, a[0], false, bm[0],
                                                    (short)0, z, false, false);
    v8f d1 = __builtin_amdgcn_wmma_f32_16x16x32_f16(false, a[1], false, bm[1],
                                                    (short)0, z, false, false);
    const int col  = t0 + mrow;
    const int roff = hiHalf * 8;
#pragma unroll
    for (int r = 0; r < 8; ++r) {
      const float v = w0 * fmaxf(d0[r], 0.0f) + w1 * fmaxf(d1[r], 0.0f);
      sc[(size_t)(roff + r) * SSTR + col] = v;
    }
  }
  __syncthreads();

  // ---- radix select: 16 threads per row, 256 elements each ----
  const int row = tid >> 4;
  const int sub = tid & 15;
  if (sub == 0) { needed[row] = TOPK; prefix[row] = 0; cntA[row] = 0; cntB[row] = 0; }
  __syncthreads();

  for (int p = 0; p < 4; ++p) {
    for (int i = tid; i < 16 * 256; i += 256) hist[i] = 0;
    __syncthreads();
    const int shift = 24 - 8 * p;
    const unsigned pref = prefix[row];
    for (int i = 0; i < 256; ++i) {
      unsigned u = __float_as_uint(sc[(size_t)row * SSTR + sub * 256 + i]);
      u = (u & 0x80000000u) ? ~u : (u | 0x80000000u);
      const bool match = (p == 0) || ((u >> (shift + 8)) == pref);
      if (match) atomicAdd(&hist[row * 256 + ((u >> shift) & 255u)], 1u);
    }
    __syncthreads();
    if (sub == 0) {
      const int need = needed[row];
      int cum = 0, dsel = 0;
      for (int dd = 255; dd >= 0; --dd) {
        const int h = (int)hist[row * 256 + dd];
        if (cum + h >= need) { dsel = dd; break; }
        cum += h;
      }
      needed[row] = need - cum;
      prefix[row] = (prefix[row] << 8) | (unsigned)dsel;
    }
    __syncthreads();
  }

  const unsigned T = prefix[row];    // exact threshold (transformed key)
  const int needEq = needed[row];
  const int nGt    = TOPK - needEq;  // exact count of keys strictly > T
  int* outp = top_idx + (rowbase + s0 + row) * TOPK;
  for (int i = 0; i < 256; ++i) {
    const int t = sub * 256 + i;
    unsigned u = __float_as_uint(sc[(size_t)row * SSTR + t]);
    u = (u & 0x80000000u) ? ~u : (u | 0x80000000u);
    if (u > T) {
      const int pos = atomicAdd(&cntA[row], 1);
      outp[pos] = t;
    } else if (u == T) {
      const int pos = atomicAdd(&cntB[row], 1);
      if (pos < needEq) outp[nGt + pos] = t;
    }
  }
}

// ---------------- Kernel 3: sparse attention via WMMA -----------------------
#define KSTR 72     // halfs per K row in LDS (64 + pad)
#define VSTR 264    // halfs per Vt row in LDS (256 + pad), Vt is [d][k]
#define CSTR 260    // floats per score row in LDS
#define PSTR 264    // halfs per P row in LDS
static constexpr int SMEM_ATTN =
    256 * KSTR * 2 + 64 * VSTR * 2 + 16 * CSTR * 4 + 16 * PSTR * 2 + 8 * 4;

__global__ void __launch_bounds__(128)
attn_kernel(const float* __restrict__ Q,
            const _Float16* __restrict__ K16, const _Float16* __restrict__ V16,
            const int* __restrict__ top_idx, float* __restrict__ out) {
  extern __shared__ unsigned char smem[];
  _Float16* Ks = (_Float16*)smem;                      // [256][KSTR]
  _Float16* Vt = Ks + 256 * KSTR;                      // [64][VSTR] transposed
  float*    sc = (float*)(Vt + 64 * VSTR);             // [16][CSTR]
  _Float16* P  = (_Float16*)(sc + 16 * CSTR);          // [16][PSTR]
  float*    inv = (float*)(P + 16 * PSTR);             // [8]

  const int tid  = threadIdx.x;
  const int lane = tid & 31;
  const int wave = tid >> 5;
  const int bs   = blockIdx.x;          // flat b*S + s
  const int b    = bs >> 12;
  const int s    = bs & (S_ - 1);
  const int* ti  = top_idx + (size_t)bs * TOPK;

  // ---- gather f16 K (row-major) and V (transposed) into LDS ----
  for (int r = tid; r < TOPK; r += 128) {
    const int gidx = ti[r];
    const _Float16* kp = K16 + ((size_t)b * S_ + gidx) * DK;
    const _Float16* vp = V16 + ((size_t)b * S_ + gidx) * DK;
#pragma unroll
    for (int j = 0; j < DK; j += 8) {
      *(v8h*)(Ks + (size_t)r * KSTR + j) = *(const v8h*)(kp + j);
      v8h vv = *(const v8h*)(vp + j);
#pragma unroll
      for (int e = 0; e < 8; ++e)
        Vt[(size_t)(j + e) * VSTR + r] = vv[e];
    }
  }

  // ---- A tiles: Q rows = heads (8 real + 8 zero pad), K-dim = d (2x32) ----
  const int hiHalf = lane >> 4;
  const int mrow   = lane & 15;
  v16h qa[2];
#pragma unroll
  for (int dc = 0; dc < 2; ++dc) {
    v16h t = {};
    if (mrow < HQ) {
      const float* qp = Q + (((size_t)b * HQ + mrow) * S_ + s) * DK +
                        dc * 32 + hiHalf * 8;
#pragma unroll
      for (int i = 0; i < 8; ++i) t[i]     = (_Float16)qp[i];
#pragma unroll
      for (int i = 0; i < 8; ++i) t[8 + i] = (_Float16)qp[16 + i];
    }
    qa[dc] = t;
  }
  __syncthreads();

  // ---- GEMM1: scores[h][k] = scale * Q.K^T, each wave does 4 n-chunks ----
  const float scale = 0.125f;   // 1/sqrt(64)
#pragma unroll
  for (int ci = 0; ci < 4; ++ci) {
    const int c    = wave * 4 + ci;       // k-chunk 0..15
    const int kcol = c * 16 + mrow;       // gathered index within 256
    v8f acc = {};
#pragma unroll
    for (int dc = 0; dc < 2; ++dc) {
      const _Float16* p = Ks + (size_t)kcol * KSTR + dc * 32 + hiHalf * 8;
      v8h lo = *(const v8h*)p;
      v8h hi = *(const v8h*)(p + 16);
      v16h bm;
#pragma unroll
      for (int i = 0; i < 8; ++i) { bm[i] = lo[i]; bm[8 + i] = hi[i]; }
      acc = __builtin_amdgcn_wmma_f32_16x16x32_f16(false, qa[dc], false, bm,
                                                   (short)0, acc, false, false);
    }
    const int roff = hiHalf * 8;
#pragma unroll
    for (int r = 0; r < 8; ++r)
      sc[(size_t)(roff + r) * CSTR + c * 16 + mrow] = acc[r] * scale;
  }
  __syncthreads();

  // ---- softmax over 256 per head; 16 threads per head ----
  const int h = tid >> 4;     // 0..7
  const int j = tid & 15;
  float mx = -1e30f;
  for (int i = 0; i < 16; ++i)
    mx = fmaxf(mx, sc[(size_t)h * CSTR + j * 16 + i]);
#pragma unroll
  for (int m = 1; m < 16; m <<= 1) mx = fmaxf(mx, __shfl_xor(mx, m, 16));
  float sum = 0.0f;
  for (int i = 0; i < 16; ++i) {
    const float e = __expf(sc[(size_t)h * CSTR + j * 16 + i] - mx);
    sum += e;
    P[(size_t)h * PSTR + j * 16 + i] = (_Float16)e;
  }
#pragma unroll
  for (int m = 1; m < 16; m <<= 1) sum += __shfl_xor(sum, m, 16);
  if (j == 0) inv[h] = 1.0f / sum;
  for (int i = 0; i < 16; ++i)                 // zero pad rows 8..15
    P[(size_t)(8 + h) * PSTR + j * 16 + i] = (_Float16)0.0f;
  __syncthreads();

  // ---- GEMM2: out[h][d] = P(16x256) x V(256x64); wave owns one d-chunk ----
  v8f acc = {};
#pragma unroll
  for (int kc = 0; kc < 8; ++kc) {
    const _Float16* pa = P + (size_t)mrow * PSTR + kc * 32 + hiHalf * 8;
    v8h alo = *(const v8h*)pa;
    v8h ahi = *(const v8h*)(pa + 16);
    v16h am;
#pragma unroll
    for (int i = 0; i < 8; ++i) { am[i] = alo[i]; am[8 + i] = ahi[i]; }
    const int dcol = wave * 16 + mrow;
    const _Float16* pb = Vt + (size_t)dcol * VSTR + kc * 32 + hiHalf * 8;
    v8h blo = *(const v8h*)pb;
    v8h bhi = *(const v8h*)(pb + 16);
    v16h bm;
#pragma unroll
    for (int i = 0; i < 8; ++i) { bm[i] = blo[i]; bm[8 + i] = bhi[i]; }
    acc = __builtin_amdgcn_wmma_f32_16x16x32_f16(false, am, false, bm,
                                                 (short)0, acc, false, false);
  }
  if (hiHalf == 0) {                       // rows 0..7 == heads 0..7
    const int dcol = wave * 16 + mrow;
    float* op = out + (size_t)bs * (HQ * DK);
#pragma unroll
    for (int r = 0; r < 8; ++r)
      op[r * DK + dcol] = acc[r] * inv[r];
  }
}

// ---------------- Host launcher --------------------------------------------
extern "C" void kernel_launch(void* const* d_in, const int* in_sizes, int n_in,
                              void* d_out, int out_size, void* d_ws, size_t ws_size,
                              hipStream_t stream) {
  (void)in_sizes; (void)n_in; (void)out_size; (void)ws_size;
  const float* x  = (const float*)d_in[0];
  const float* Q  = (const float*)d_in[1];
  const float* K  = (const float*)d_in[2];
  const float* V  = (const float*)d_in[3];
  const float* Wq = (const float*)d_in[4];
  const float* bq = (const float*)d_in[5];
  const float* Wk = (const float*)d_in[6];
  const float* bk = (const float*)d_in[7];
  const float* g  = (const float*)d_in[8];
  const float* bl = (const float*)d_in[9];
  const float* iw = (const float*)d_in[10];
  float* out = (float*)d_out;

  _Float16* qi  = (_Float16*)d_ws;                       // 1 MB
  _Float16* ki  = qi + (size_t)B_ * S_ * 64;             // 1 MB
  int*      tki = (int*)(ki + (size_t)B_ * S_ * 64);     // 8 MB
  _Float16* K16 = (_Float16*)(tki + (size_t)B_ * S_ * TOPK); // 1 MB
  _Float16* V16 = K16 + (size_t)B_ * S_ * DK;                // 1 MB

  hipFuncSetAttribute(reinterpret_cast<const void*>(score_topk_kernel),
                      hipFuncAttributeMaxDynamicSharedMemorySize, SMEM_SCORE);
  hipFuncSetAttribute(reinterpret_cast<const void*>(attn_kernel),
                      hipFuncAttributeMaxDynamicSharedMemorySize, SMEM_ATTN);

  proj_ln_kernel<<<B_ * S_ / 8, 128, 0, stream>>>(x, Wq, bq, Wk, bk, g, bl, qi, ki);
  cvt_kv_kernel<<<(B_ * S_ * DK) / (256 * 8), 256, 0, stream>>>(K, V, K16, V16);
  score_topk_kernel<<<B_ * S_ / 16, 256, SMEM_SCORE, stream>>>(qi, ki, iw, tki);
  attn_kernel<<<B_ * S_, 128, SMEM_ATTN, stream>>>(Q, K16, V16, tki, out);
}
